// EMBlock_45174466019395
// MI455X (gfx1250) — matrix operand order repaired and checked by reading, not compiled, not run on hardware
//
#include <hip/hip_runtime.h>
#include <stdint.h>

// ---------------------------------------------------------------------------
// Problem constants (fixed by the reference): B=2, L=2048, D=1024, H=16, hd=64
// ---------------------------------------------------------------------------

typedef __attribute__((ext_vector_type(16))) __bf16 v16bf;
typedef __attribute__((ext_vector_type(8)))  float  v8f;

#define DEV __device__ __forceinline__

DEV v8f wmma_bf16(v16bf a, v16bf b, v8f c) {
  // D = A(16x32 bf16) * B(32x16 bf16) + C(16x16 f32)
  return __builtin_amdgcn_wmma_f32_16x16x32_bf16(
      /*neg_a=*/false, a, /*neg_b=*/false, b,
      /*c_mod=*/(short)0, c, /*reuse_a=*/false, /*reuse_b=*/false);
}

union FragU { uint4 u[2]; v16bf v; };

// A fragment (16x32, 16-bit): lane holds row M = lane%16.
// lanes 0-15: K = 0..7 then 16..23 ; lanes 16-31: K = 8..15 then 24..31.
DEV v16bf load_frag_a(const __bf16* row32, int lane) {
  const int kh8 = (lane & 16) ? 8 : 0;
  FragU f;
  f.u[0] = *(const uint4*)(row32 + kh8);
  f.u[1] = *(const uint4*)(row32 + 16 + kh8);
  return f.v;
}

// B fragment (32x16, 16-bit): lane holds column N = lane%16.
// lanes 0-15: K = 0..7 then 16..23 ; lanes 16-31: K = 8..15 then 24..31.
DEV v16bf load_frag_b(const __bf16* row32, int lane) {
  const int off = (lane & 16) ? 8 : 0;
  FragU f;
  f.u[0] = *(const uint4*)(row32 + off);
  f.u[1] = *(const uint4*)(row32 + 16 + off);
  return f.v;
}

// ---------------------------------------------------------------------------
// Tensor Data Mover staging (gfx1250).  Descriptor per CDNA5 ISA ch.8:
//   group0: [count=1][lds_addr][global_addr lo][global_addr hi | type=2<<30]
//   group1: data_size(2B), tensor_dim0/1 == tile_dim0/1 (tiles never OOB),
//           tensor_dim0_stride = row stride in elements.
// Issued by one wave; completion via TENSORcnt.
// This toolchain exposes the 6-arg builtin (extra int32x8 group, zero-filled
// for <=2D tensors).
// ---------------------------------------------------------------------------
#if __has_builtin(__builtin_amdgcn_tensor_load_to_lds) && \
    __has_builtin(__builtin_amdgcn_s_wait_tensorcnt)
#define USE_TDM 1
#else
#define USE_TDM 0
#endif

#if USE_TDM
typedef unsigned int u32x4 __attribute__((ext_vector_type(4)));
typedef int          i32x4 __attribute__((ext_vector_type(4)));
typedef int          i32x8 __attribute__((ext_vector_type(8)));

DEV uint32_t lds_addr32(const void* p) {
  return (uint32_t)(size_t)(__attribute__((address_space(3))) const char*)p;
}

// 2D tile of bf16: tile_w elems per row, tile_h rows, global row stride
// row_stride elems; packed contiguously into LDS at ldsoff.
DEV void tdm_load_2d(uint64_t gaddr, uint32_t ldsoff, uint32_t tile_w,
                     uint32_t tile_h, uint32_t row_stride) {
  u32x4 g0;
  g0[0] = 1u;                                           // count=1, user mode
  g0[1] = ldsoff;                                       // lds_addr
  g0[2] = (uint32_t)gaddr;                              // global_addr[31:0]
  g0[3] = (uint32_t)((gaddr >> 32) & 0x1FFFFFFu) | (2u << 30);  // [56:32]|type
  i32x8 g1;
  g1[0] = (int)0x00010000u;                             // data_size=1 -> 2B
  g1[1] = (int)((tile_w & 0xffffu) << 16);              // tensor_dim0 lo
  g1[2] = (int)((tile_w >> 16) | ((tile_h & 0xffffu) << 16));  // td0 hi|td1 lo
  g1[3] = (int)((tile_h >> 16) | ((tile_w & 0xffffu) << 16));  // td1 hi|tile_dim0
  g1[4] = (int)(tile_h & 0xffffu);                      // tile_dim1 (tile_dim2=0)
  g1[5] = (int)row_stride;                              // tensor_dim0_stride lo
  g1[6] = 0;                                            // stride hi / dim1_stride lo
  g1[7] = 0;
  i32x4 z4 = {0, 0, 0, 0};
  i32x8 z8 = {0, 0, 0, 0, 0, 0, 0, 0};
  __builtin_amdgcn_tensor_load_to_lds(g0, g1, z4, z4, z8, 0);
}
#endif  // USE_TDM

// ---------------------------------------------------------------------------
// Prep kernels
// ---------------------------------------------------------------------------

__global__ void k_f32_to_bf16(const float* __restrict__ src,
                              __bf16* __restrict__ dst, int n) {
  int i = (blockIdx.x * 256 + threadIdx.x) * 4;
  if (i + 3 < n) {
    float4 f = *(const float4*)(src + i);
    dst[i + 0] = (__bf16)f.x;
    dst[i + 1] = (__bf16)f.y;
    dst[i + 2] = (__bf16)f.z;
    dst[i + 3] = (__bf16)f.w;
  }
}

// src[K][N] f32 -> dst[N][K] bf16
__global__ void k_transpose_bf16(const float* __restrict__ src,
                                 __bf16* __restrict__ dst, int K, int N) {
  size_t i = (size_t)blockIdx.x * 256 + threadIdx.x;  // dst-linear
  int k = (int)(i % K);
  int n = (int)(i / K);
  dst[i] = (__bf16)src[(size_t)k * N + n];
}

// src [B,L,H*hd] f32 -> dst [B,H,L,hd] bf16   (gate streams q_g / k_g)
__global__ void k_reshape_heads(const float* __restrict__ src,
                                __bf16* __restrict__ dst) {
  int i = blockIdx.x * 256 + threadIdx.x;  // dst-linear, 2*16*2048*64 total
  int d  = i & 63;
  int l  = (i >> 6) & 2047;
  int h2 = (i >> 17) & 15;
  int b2 = i >> 21;
  dst[i] = (__bf16)src[(((size_t)b2 * 2048 + l) << 10) + h2 * 64 + d];
}

// ---------------------------------------------------------------------------
// GEMM1: qkv = Xbf[4096,1024] @ WqkvT[3072,1024]^T, scatter into head-major
// q/k/v bf16 [B,H,L,hd].  256 thr = 8 waves, tile 128x128x32, ping-pong LDS,
// TDM-staged (wave0 issues DMA for tile k+1 while all waves run WMMAs on k).
// ---------------------------------------------------------------------------

__global__ __launch_bounds__(256)
void k_gemm_qkv(const __bf16* __restrict__ A,   // [4096,1024]
                const __bf16* __restrict__ Bt,  // [3072,1024]
                __bf16* __restrict__ qh, __bf16* __restrict__ kh,
                __bf16* __restrict__ vh) {
  __shared__ __bf16 As[2][128 * 32];
  __shared__ __bf16 Bs[2][128 * 32];
  const int tid = threadIdx.x, lane = tid & 31, wave = tid >> 5;
  const int wm = wave & 3, wn = wave >> 2;
  const int bm = blockIdx.x * 128, bn = blockIdx.y * 128;

  v8f z = {0.f, 0.f, 0.f, 0.f, 0.f, 0.f, 0.f, 0.f};
  v8f acc[8];
  for (int i = 0; i < 8; i++) acc[i] = z;

#if USE_TDM
  const uint64_t Ab = (uint64_t)(size_t)(A + (size_t)bm * 1024);
  const uint64_t Bb = (uint64_t)(size_t)(Bt + (size_t)bn * 1024);
  if (wave == 0) {
    tdm_load_2d(Ab, lds_addr32(As[0]), 32, 128, 1024);
    tdm_load_2d(Bb, lds_addr32(Bs[0]), 32, 128, 1024);
    __builtin_amdgcn_s_wait_tensorcnt(0);
  }
#else
  for (int s = 0; s < 2; s++) {
    int slot = tid + s * 256;
    int row = slot >> 2, seg = slot & 3;
    *(uint4*)(As[0] + row * 32 + seg * 8) =
        *(const uint4*)(A + (size_t)(bm + row) * 1024 + seg * 8);
    *(uint4*)(Bs[0] + row * 32 + seg * 8) =
        *(const uint4*)(Bt + (size_t)(bn + row) * 1024 + seg * 8);
  }
#endif
  __syncthreads();

  int cur = 0;
  for (int it = 0; it < 32; ++it) {
    const bool more = (it + 1 < 32);
#if USE_TDM
    if (more && wave == 0) {                 // DMA next tile during compute
      uint32_t kb = (uint32_t)(it + 1) * 64; // 32 elems * 2B
      tdm_load_2d(Ab + kb, lds_addr32(As[cur ^ 1]), 32, 128, 1024);
      tdm_load_2d(Bb + kb, lds_addr32(Bs[cur ^ 1]), 32, 128, 1024);
    }
#else
    uint4 na[2], nb[2];
    if (more) {
      int kkn = (it + 1) * 32;
      for (int s = 0; s < 2; s++) {
        int slot = tid + s * 256;
        int row = slot >> 2, seg = slot & 3;
        na[s] = *(const uint4*)(A + (size_t)(bm + row) * 1024 + kkn + seg * 8);
        nb[s] = *(const uint4*)(Bt + (size_t)(bn + row) * 1024 + kkn + seg * 8);
      }
    }
#endif
    v16bf af[2], bfr[4];
    for (int t = 0; t < 2; t++)
      af[t] = load_frag_a(As[cur] + (wm * 32 + t * 16 + (lane & 15)) * 32, lane);
    for (int u = 0; u < 4; u++)
      bfr[u] = load_frag_b(Bs[cur] + (wn * 64 + u * 16 + (lane & 15)) * 32, lane);
    for (int t = 0; t < 2; t++)
      for (int u = 0; u < 4; u++)
        acc[t * 4 + u] = wmma_bf16(af[t], bfr[u], acc[t * 4 + u]);

    if (more) {
#if USE_TDM
      if (wave == 0) __builtin_amdgcn_s_wait_tensorcnt(0);
#else
      for (int s = 0; s < 2; s++) {
        int slot = tid + s * 256;
        int row = slot >> 2, seg = slot & 3;
        *(uint4*)(As[cur ^ 1] + row * 32 + seg * 8) = na[s];
        *(uint4*)(Bs[cur ^ 1] + row * 32 + seg * 8) = nb[s];
      }
#endif
      __syncthreads();
      cur ^= 1;
    }
  }

  // Epilogue: C elem r <-> M = r + 8*(lane>=16), N = lane%16.
  const int mh8 = (lane & 16) ? 8 : 0;
  for (int t = 0; t < 2; t++)
    for (int u = 0; u < 4; u++) {
      int N3 = bn + wn * 64 + u * 16 + (lane & 15);
      int stream = N3 >> 10, rem = N3 & 1023, hh = rem >> 6, dd = rem & 63;
      __bf16* dst = (stream == 0) ? qh : (stream == 1) ? kh : vh;
      for (int r = 0; r < 8; r++) {
        int M = bm + wm * 32 + t * 16 + r + mh8;
        int bb = M >> 11, ll = M & 2047;
        dst[(((size_t)(bb * 16 + hh)) * 2048 + ll) * 64 + dd] =
            (__bf16)acc[t * 4 + u][r];
      }
    }
}

// ---------------------------------------------------------------------------
// Fused dual-stream causal flash attention.
// Grid: B*H*(L/64) blocks, 128 threads = 4 waves; wave owns 16 q-rows.
// Per 32-key step: TDM-stage Kx/Kg (4KB contiguous each, wave0 issues),
// manual V-transpose scatter, 8 WMMA for S, online softmax, 4 WMMA for P@V.
// ---------------------------------------------------------------------------

__global__ __launch_bounds__(128)
void k_attn(const __bf16* __restrict__ qh, const __bf16* __restrict__ kh,
            const __bf16* __restrict__ vh, const __bf16* __restrict__ qgh,
            const __bf16* __restrict__ kgh, __bf16* __restrict__ ybf) {
  __shared__ __bf16 kxs[32 * 64];      // K tile (x stream)   [key][d]
  __shared__ __bf16 kgs[32 * 64];      // K tile (gate)       [key][d]
  __shared__ __bf16 vts[64 * 32];      // V tile transposed   [d][key]
  __shared__ __bf16 ps[4 * 16 * 32];   // per-wave P re-swizzle scratch

  const int tid = threadIdx.x, lane = tid & 31, wave = tid >> 5;
  const int qt = blockIdx.x & 31, bh = blockIdx.x >> 5;
  const int bb = bh >> 4, hh = bh & 15;
  const int q0w = qt * 64 + wave * 16;
  // combined scale: (1/sqrt(64)) / (log(2048) * T)
  const float SC  = 0.125f / 7.624618986159398f;
  const float NEG = -__builtin_inff();

  const __bf16* qrow  = qh  + ((size_t)bh * 2048 + q0w + (lane & 15)) * 64;
  const __bf16* qgrow = qgh + ((size_t)bh * 2048 + q0w + (lane & 15)) * 64;
  v16bf qx0 = load_frag_a(qrow, lane),  qx1 = load_frag_a(qrow + 32, lane);
  v16bf qg0 = load_frag_a(qgrow, lane), qg1 = load_frag_a(qgrow + 32, lane);

  float m_run[8], l_run[8];
  v8f z = {0.f, 0.f, 0.f, 0.f, 0.f, 0.f, 0.f, 0.f};
  v8f acc[4];
  for (int r = 0; r < 8; r++) { m_run[r] = NEG; l_run[r] = 0.f; }
  for (int u = 0; u < 4; u++) acc[u] = z;
  const int mh8 = (lane & 16) ? 8 : 0;
  __bf16* pw = ps + wave * 512;

  const int nkt = (qt + 1) * 2;                    // causal: keys <= qt*64+63
  for (int kt = 0; kt < nkt; ++kt) {
    const size_t krow0 = (size_t)bh * 2048 + kt * 32;
#if USE_TDM
    if (wave == 0) {                               // 4KB contiguous DMA each
      tdm_load_2d((uint64_t)(size_t)(kh  + krow0 * 64), lds_addr32(kxs),
                  2048, 1, 2048);
      tdm_load_2d((uint64_t)(size_t)(kgh + krow0 * 64), lds_addr32(kgs),
                  2048, 1, 2048);
    }
#endif
    // V staged manually (needs transpose into [d][key])
    for (int s = 0; s < 2; s++) {
      int slot = tid + s * 128;                    // 256 slots = 32 rows x 8
      int row = slot >> 3, seg = slot & 7;
      size_t g = (krow0 + row) * 64 + seg * 8;
      union { uint4 u; __bf16 e[8]; } vv;
      vv.u = *(const uint4*)(vh + g);
      for (int e = 0; e < 8; e++) vts[(seg * 8 + e) * 32 + row] = vv.e[e];
#if !USE_TDM
      *(uint4*)(kxs + row * 64 + seg * 8) = *(const uint4*)(kh + g);
      *(uint4*)(kgs + row * 64 + seg * 8) = *(const uint4*)(kgh + g);
#endif
    }
#if USE_TDM
    if (wave == 0) __builtin_amdgcn_s_wait_tensorcnt(0);
#endif
    __syncthreads();

    // --- scores: S = (q kx^T + qg kg^T), two 16-key C tiles ----------------
    float vals[2][8];
    for (int half = 0; half < 2; ++half) {
      const __bf16* kxr = kxs + (half * 16 + (lane & 15)) * 64;
      const __bf16* kgr = kgs + (half * 16 + (lane & 15)) * 64;
      v8f s = z;
      s = wmma_bf16(qx0, load_frag_b(kxr, lane), s);
      s = wmma_bf16(qx1, load_frag_b(kxr + 32, lane), s);
      s = wmma_bf16(qg0, load_frag_b(kgr, lane), s);
      s = wmma_bf16(qg1, load_frag_b(kgr + 32, lane), s);
      int keyg = kt * 32 + half * 16 + (lane & 15);
      for (int r = 0; r < 8; r++) {
        int rowg = q0w + r + mh8;
        vals[half][r] = (keyg <= rowg) ? s[r] * SC : NEG;
      }
    }

    // --- online softmax (row stats via 16-lane shuffle reductions) ---------
    for (int r = 0; r < 8; r++) {
      float mx = fmaxf(vals[0][r], vals[1][r]);
      mx = fmaxf(mx, __shfl_xor(mx, 1, 32));
      mx = fmaxf(mx, __shfl_xor(mx, 2, 32));
      mx = fmaxf(mx, __shfl_xor(mx, 4, 32));
      mx = fmaxf(mx, __shfl_xor(mx, 8, 32));
      float mn = fmaxf(m_run[r], mx);
      bool ninf = (mn == NEG);
      float corr = ninf ? 1.f : __expf(m_run[r] - mn);
      float p0 = ninf ? 0.f : __expf(vals[0][r] - mn);
      float p1 = ninf ? 0.f : __expf(vals[1][r] - mn);
      float rs = p0 + p1;
      rs += __shfl_xor(rs, 1, 32);
      rs += __shfl_xor(rs, 2, 32);
      rs += __shfl_xor(rs, 4, 32);
      rs += __shfl_xor(rs, 8, 32);
      l_run[r] = l_run[r] * corr + rs;
      m_run[r] = mn;
      int ml = r + mh8, col = lane & 15;
      pw[ml * 32 + col]      = (__bf16)p0;          // re-swizzle P via LDS
      pw[ml * 32 + 16 + col] = (__bf16)p1;
      for (int u = 0; u < 4; u++) acc[u][r] *= corr;
    }
    asm volatile("s_wait_dscnt 0" ::: "memory");     // intra-wave DS RAW

    // --- y += P(16x32) @ V(32x64) ------------------------------------------
    v16bf pA = load_frag_a(pw + (lane & 15) * 32, lane);
    for (int u = 0; u < 4; u++) {
      v16bf vB = load_frag_b(vts + (u * 16 + (lane & 15)) * 32, lane);
      acc[u] = wmma_bf16(pA, vB, acc[u]);
    }
    __syncthreads();
  }

  // --- finalize: y /= l, store bf16 [B,L,D] --------------------------------
  for (int r = 0; r < 8; r++) {
    float inv = 1.f / l_run[r];
    int rowg = q0w + r + mh8;
    size_t base = ((size_t)(bb * 2048 + rowg)) * 1024 + hh * 64;
    for (int u = 0; u < 4; u++)
      ybf[base + u * 16 + (lane & 15)] = (__bf16)(acc[u][r] * inv);
  }
}

// ---------------------------------------------------------------------------
// GEMM2: out[4096,1024] f32 = ybf[4096,1024] @ WoutT[1024,1024]^T
// Same TDM double-buffered pipeline as GEMM1.
// ---------------------------------------------------------------------------

__global__ __launch_bounds__(256)
void k_gemm_out(const __bf16* __restrict__ A,   // [4096,1024]
                const __bf16* __restrict__ Bt,  // [1024,1024]
                float* __restrict__ out) {
  __shared__ __bf16 As[2][128 * 32];
  __shared__ __bf16 Bs[2][128 * 32];
  const int tid = threadIdx.x, lane = tid & 31, wave = tid >> 5;
  const int wm = wave & 3, wn = wave >> 2;
  const int bm = blockIdx.x * 128, bn = blockIdx.y * 128;

  v8f z = {0.f, 0.f, 0.f, 0.f, 0.f, 0.f, 0.f, 0.f};
  v8f acc[8];
  for (int i = 0; i < 8; i++) acc[i] = z;

#if USE_TDM
  const uint64_t Ab = (uint64_t)(size_t)(A + (size_t)bm * 1024);
  const uint64_t Bb = (uint64_t)(size_t)(Bt + (size_t)bn * 1024);
  if (wave == 0) {
    tdm_load_2d(Ab, lds_addr32(As[0]), 32, 128, 1024);
    tdm_load_2d(Bb, lds_addr32(Bs[0]), 32, 128, 1024);
    __builtin_amdgcn_s_wait_tensorcnt(0);
  }
#else
  for (int s = 0; s < 2; s++) {
    int slot = tid + s * 256;
    int row = slot >> 2, seg = slot & 3;
    *(uint4*)(As[0] + row * 32 + seg * 8) =
        *(const uint4*)(A + (size_t)(bm + row) * 1024 + seg * 8);
    *(uint4*)(Bs[0] + row * 32 + seg * 8) =
        *(const uint4*)(Bt + (size_t)(bn + row) * 1024 + seg * 8);
  }
#endif
  __syncthreads();

  int cur = 0;
  for (int it = 0; it < 32; ++it) {
    const bool more = (it + 1 < 32);
#if USE_TDM
    if (more && wave == 0) {
      uint32_t kb = (uint32_t)(it + 1) * 64;
      tdm_load_2d(Ab + kb, lds_addr32(As[cur ^ 1]), 32, 128, 1024);
      tdm_load_2d(Bb + kb, lds_addr32(Bs[cur ^ 1]), 32, 128, 1024);
    }
#else
    uint4 na[2], nb[2];
    if (more) {
      int kkn = (it + 1) * 32;
      for (int s = 0; s < 2; s++) {
        int slot = tid + s * 256;
        int row = slot >> 2, seg = slot & 3;
        na[s] = *(const uint4*)(A + (size_t)(bm + row) * 1024 + kkn + seg * 8);
        nb[s] = *(const uint4*)(Bt + (size_t)(bn + row) * 1024 + kkn + seg * 8);
      }
    }
#endif
    v16bf af[2], bfr[4];
    for (int t = 0; t < 2; t++)
      af[t] = load_frag_a(As[cur] + (wm * 32 + t * 16 + (lane & 15)) * 32, lane);
    for (int u = 0; u < 4; u++)
      bfr[u] = load_frag_b(Bs[cur] + (wn * 64 + u * 16 + (lane & 15)) * 32, lane);
    for (int t = 0; t < 2; t++)
      for (int u = 0; u < 4; u++)
        acc[t * 4 + u] = wmma_bf16(af[t], bfr[u], acc[t * 4 + u]);

    if (more) {
#if USE_TDM
      if (wave == 0) __builtin_amdgcn_s_wait_tensorcnt(0);
#else
      for (int s = 0; s < 2; s++) {
        int slot = tid + s * 256;
        int row = slot >> 2, seg = slot & 3;
        *(uint4*)(As[cur ^ 1] + row * 32 + seg * 8) = na[s];
        *(uint4*)(Bs[cur ^ 1] + row * 32 + seg * 8) = nb[s];
      }
#endif
      __syncthreads();
      cur ^= 1;
    }
  }

  const int mh8 = (lane & 16) ? 8 : 0;
  for (int t = 0; t < 2; t++)
    for (int u = 0; u < 4; u++) {
      int N = bn + wn * 64 + u * 16 + (lane & 15);
      for (int r = 0; r < 8; r++) {
        int M = bm + wm * 32 + t * 16 + r + mh8;
        out[(size_t)M * 1024 + N] = acc[t * 4 + u][r];
      }
    }
}

// ---------------------------------------------------------------------------
// Launch
// ---------------------------------------------------------------------------

extern "C" void kernel_launch(void* const* d_in, const int* in_sizes, int n_in,
                              void* d_out, int out_size, void* d_ws,
                              size_t ws_size, hipStream_t stream) {
  const float* x    = (const float*)d_in[0];
  const float* q_g  = (const float*)d_in[1];
  const float* k_g  = (const float*)d_in[2];
  const float* Wqkv = (const float*)d_in[3];
  const float* Wout = (const float*)d_in[4];
  float* out = (float*)d_out;

  // Workspace carve (bf16 intermediates, ~67 MB total, fits L2 easily)
  uint8_t* p = (uint8_t*)d_ws;
  auto carve = [&](size_t elems) {
    __bf16* r = (__bf16*)p;
    p += (elems * 2 + 255) & ~(size_t)255;
    return r;
  };
  __bf16* xbf   = carve((size_t)4096 * 1024);
  __bf16* wqkvT = carve((size_t)3072 * 1024);
  __bf16* woutT = carve((size_t)1024 * 1024);
  __bf16* qh    = carve((size_t)32 * 2048 * 64);
  __bf16* kh    = carve((size_t)32 * 2048 * 64);
  __bf16* vh    = carve((size_t)32 * 2048 * 64);
  __bf16* qgh   = carve((size_t)32 * 2048 * 64);
  __bf16* kgh   = carve((size_t)32 * 2048 * 64);
  __bf16* ybf   = carve((size_t)4096 * 1024);

  k_f32_to_bf16<<<(4096 * 1024) / 1024, 256, 0, stream>>>(x, xbf, 4096 * 1024);
  k_transpose_bf16<<<(1024 * 3072) / 256, 256, 0, stream>>>(Wqkv, wqkvT, 1024, 3072);
  k_transpose_bf16<<<(1024 * 1024) / 256, 256, 0, stream>>>(Wout, woutT, 1024, 1024);
  k_reshape_heads<<<(2 * 2048 * 1024) / 256, 256, 0, stream>>>(q_g, qgh);
  k_reshape_heads<<<(2 * 2048 * 1024) / 256, 256, 0, stream>>>(k_g, kgh);

  dim3 g1(32, 24);
  k_gemm_qkv<<<g1, 256, 0, stream>>>(xbf, wqkvT, qh, kh, vh);

  k_attn<<<2 * 16 * 32, 128, 0, stream>>>(qh, kh, vh, qgh, kgh, ybf);

  dim3 g2(32, 8);
  k_gemm_out<<<g2, 256, 0, stream>>>(ybf, woutT, out);

  (void)in_sizes; (void)n_in; (void)out_size; (void)ws_size;
}
